// QuantizedMLP_30940944400880
// MI455X (gfx1250) — matrix operand order repaired and checked by reading, not compiled, not run on hardware
//
#include <hip/hip_runtime.h>
#include <hip/hip_bf16.h>

typedef __bf16 bf16_t;
typedef __attribute__((ext_vector_type(16))) __bf16 v16bf;
typedef __attribute__((ext_vector_type(8)))  __bf16 v8bf;
typedef __attribute__((ext_vector_type(8)))  float   v8f;

// ---------------------------------------------------------------------------
// Per-row symmetric fake-quant: wq = clamp(round(w/scale),-128,127) as bf16
// (exact: |q|<=128 fits bf16 significand), scale[row] = max(amax/127, 1e-12).
// ---------------------------------------------------------------------------
__global__ __launch_bounds__(256)
void quant_rows_kernel(const float* __restrict__ w, bf16_t* __restrict__ wq,
                       float* __restrict__ scale, int K) {
  __shared__ float red[256];
  __shared__ float s_scale;
  const int row = blockIdx.x;
  const float* wr = w + (size_t)row * K;
  float amax = 0.f;
  for (int k = threadIdx.x; k < K; k += 256) amax = fmaxf(amax, fabsf(wr[k]));
  red[threadIdx.x] = amax;
  __syncthreads();
  for (int s = 128; s > 0; s >>= 1) {
    if (threadIdx.x < s) red[threadIdx.x] = fmaxf(red[threadIdx.x], red[threadIdx.x + s]);
    __syncthreads();
  }
  if (threadIdx.x == 0) {
    float sc = fmaxf(red[0] * (1.0f / 127.0f), 1e-12f);
    s_scale = sc;
    scale[row] = sc;
  }
  __syncthreads();
  const float inv = 1.0f / s_scale;
  bf16_t* wqr = wq + (size_t)row * K;
  for (int k = threadIdx.x; k < K; k += 256) {
    float q = rintf(wr[k] * inv);              // round-half-even == jnp.round
    q = fminf(fmaxf(q, -128.0f), 127.0f);
    wqr[k] = (bf16_t)q;                        // exact
  }
}

// ---------------------------------------------------------------------------
// heads_w [1000,1024] f32 -> bf16 padded to [1024,1024] (pad rows = 0)
// ---------------------------------------------------------------------------
__global__ __launch_bounds__(256)
void heads_convert_kernel(const float* __restrict__ hw, bf16_t* __restrict__ hwq,
                          int n_valid, int K) {
  const int row = blockIdx.x;
  bf16_t* dst = hwq + (size_t)row * K;
  if (row < n_valid) {
    const float* src = hw + (size_t)row * K;
    for (int k = threadIdx.x; k < K; k += 256) dst[k] = (bf16_t)src[k];
  } else {
    for (int k = threadIdx.x; k < K; k += 256) dst[k] = (bf16_t)0.0f;
  }
}

// ---------------------------------------------------------------------------
// C[m,n] = sum_k A[m,k] * W[n,k]   (W row-major == WMMA B column layout)
// M_TILE=64 rows/WG, full N=1024 covered by 8 waves * 128 cols.
// Per wave per K-step(32): 4 M-frags x 8 N-frags = 32 v_wmma vs 16 B-loads.
// K is a template constant so each B address is SGPR base + 32-bit lane
// offset + compile-time immediate (nf*16*K*2 <= 688KB < 24-bit IOFFSET).
// Double-buffered LDS A tile -> one barrier per K-step.
// ---------------------------------------------------------------------------
template <int K, bool A_F32, bool HAS_SCALE, bool RELU, bool OUT_BF16>
__global__ __launch_bounds__(256)
void gemm_nt_kernel(const void* __restrict__ Aptr, const bf16_t* __restrict__ W,
                    const float* __restrict__ scale, const float* __restrict__ bias,
                    void* __restrict__ Out, int n_valid) {
  constexpr int NCOLS = 1024;
  constexpr int MT = 64;
  constexpr int LDSROW = 40;                  // 80B row stride, 16B-aligned chunks
  constexpr int BUFE = MT * LDSROW;
  __shared__ bf16_t ldsA[2 * BUFE];

  const int tid  = threadIdx.x;
  const int lane = tid & 31;
  const int wave = tid >> 5;
  const size_t m0 = (size_t)blockIdx.x * MT;
  const int wn0 = wave * 128;

  // coop A loader: 64 rows x 32 cols, 8 elems/thread: row=tid/4, col=(tid%4)*8
  const int ldr = tid >> 2;
  const int ldc = (tid & 3) * 8;

  // A-fragment addressing in contiguous k order: element s of lane half h is k = 16h + s (B matches)
  const int arow = lane & 15;
  const int ac0  = (lane < 16) ? 0 : 16;      // chunks {ac0, ac0+8}
  const int bkh  = (lane >> 4) * 16;          // B-fragment K-half

  // B: uniform SGPR base + per-lane 32-bit byte offset (+64/step), imm per nf
  const char* Wb = (const char*)W;
  unsigned boff = (unsigned)(((wn0 + (lane & 15)) * K + bkh) * 2);

  v8f zero = {0.f, 0.f, 0.f, 0.f, 0.f, 0.f, 0.f, 0.f};
  v8f acc[4][8];
#pragma unroll
  for (int i = 0; i < 4; ++i)
#pragma unroll
    for (int j = 0; j < 8; ++j) acc[i][j] = zero;

  auto load_tile = [&](int k0, int buf) {
    bf16_t* dst = &ldsA[buf * BUFE + ldr * LDSROW + ldc];
    if (A_F32) {
      const float* A = (const float*)Aptr + (m0 + ldr) * (size_t)K + k0 + ldc;
      const float4 v0 = *(const float4*)A;
      const float4 v1 = *(const float4*)(A + 4);
      v8bf t;
      t[0] = (bf16_t)v0.x; t[1] = (bf16_t)v0.y; t[2] = (bf16_t)v0.z; t[3] = (bf16_t)v0.w;
      t[4] = (bf16_t)v1.x; t[5] = (bf16_t)v1.y; t[6] = (bf16_t)v1.z; t[7] = (bf16_t)v1.w;
      *(v8bf*)dst = t;                        // one ds_store_b128
    } else {
      const bf16_t* A = (const bf16_t*)Aptr + (m0 + ldr) * (size_t)K + k0 + ldc;
      *(uint4*)dst = *(const uint4*)A;        // 16B global -> 16B LDS
    }
  };

  load_tile(0, 0);
  __syncthreads();

  for (int k0 = 0; k0 < K; k0 += 32) {
    const int buf = (k0 >> 5) & 1;
    if (k0 + 32 < K) load_tile(k0 + 32, buf ^ 1);  // fill other buffer

    // ---- 4 A fragments from LDS (two b128 reads each) ----
    v16bf afr[4];
#pragma unroll
    for (int mi = 0; mi < 4; ++mi) {
      const int base = buf * BUFE + (mi * 16 + arow) * LDSROW + ac0;
      v8bf lo = *(const v8bf*)&ldsA[base];
      v8bf hi = *(const v8bf*)&ldsA[base + 8];
#pragma unroll
      for (int e = 0; e < 8; ++e) { afr[mi][e] = lo[e]; afr[mi][8 + e] = hi[e]; }
    }

    // ---- 8 N-frags x 4 M-frags WMMA; B from L2-resident W ----
#pragma unroll
    for (int nf = 0; nf < 8; ++nf) {
      const v16bf bfr = *(const v16bf*)(Wb + boff + (size_t)(nf * 16 * K * 2));
#pragma unroll
      for (int mi = 0; mi < 4; ++mi) {
        acc[mi][nf] = __builtin_amdgcn_wmma_f32_16x16x32_bf16(
            false, afr[mi], false, bfr, (short)0, acc[mi][nf], false, false);
      }
    }
    boff += 64;                               // advance K by 32 bf16 elements
    __syncthreads();
  }

  // ---- epilogue: scale/bias/relu in C/D layout; store bf16 or guarded f32 ----
  const int cn = lane & 15;
  const int rh = (lane >> 4) * 8;
#pragma unroll
  for (int nf = 0; nf < 8; ++nf) {
    const int col = wn0 + nf * 16 + cn;
    float sc = 1.0f, bs = 0.0f;
    if (HAS_SCALE) sc = scale[col];
    if (col < n_valid) bs = bias[col];
#pragma unroll
    for (int mi = 0; mi < 4; ++mi) {
#pragma unroll
      for (int r = 0; r < 8; ++r) {
        float v = fmaf(acc[mi][nf][r], sc, bs);
        if (RELU) v = fmaxf(v, 0.0f);
        const size_t row = m0 + mi * 16 + rh + r;
        if (OUT_BF16) {
          ((bf16_t*)Out)[row * NCOLS + col] = (bf16_t)v;
        } else {
          if (col < n_valid) ((float*)Out)[row * (size_t)n_valid + col] = v;
        }
      }
    }
  }
}

// ---------------------------------------------------------------------------
extern "C" void kernel_launch(void* const* d_in, const int* in_sizes, int n_in,
                              void* d_out, int out_size, void* d_ws, size_t ws_size,
                              hipStream_t stream) {
  const int B = 16384, D_IN = 3072, H = 1024, NC = 1000;

  const float* x  = (const float*)d_in[0];
  const float* w1 = (const float*)d_in[1];
  const float* b1 = (const float*)d_in[2];
  const float* w2 = (const float*)d_in[3];
  const float* b2 = (const float*)d_in[4];
  const float* hw = (const float*)d_in[5];
  const float* hb = (const float*)d_in[6];

  char* ws = (char*)d_ws;
  bf16_t* w1q = (bf16_t*)ws;            ws += (size_t)H * D_IN * 2;   // 6.29 MB
  float*  s1  = (float*)ws;             ws += 4096;
  bf16_t* w2q = (bf16_t*)ws;            ws += (size_t)H * H * 2;      // 2.10 MB
  float*  s2  = (float*)ws;             ws += 4096;
  bf16_t* hwq = (bf16_t*)ws;            ws += (size_t)1024 * H * 2;   // 2.10 MB (padded)
  bf16_t* h1  = (bf16_t*)ws;            ws += (size_t)B * H * 2;      // 33.6 MB
  bf16_t* h2  = (bf16_t*)ws;            ws += (size_t)B * H * 2;      // 33.6 MB

  // weight prep (tiny, L2-resident afterwards)
  quant_rows_kernel<<<H, 256, 0, stream>>>(w1, w1q, s1, D_IN);
  quant_rows_kernel<<<H, 256, 0, stream>>>(w2, w2q, s2, H);
  heads_convert_kernel<<<1024, 256, 0, stream>>>(hw, hwq, NC, H);

  // fc1: x f32 -> bf16-in-LDS, WMMA, scale+bias+relu, bf16 out
  gemm_nt_kernel<3072, true,  true,  true,  true ><<<B / 64, 256, 0, stream>>>(
      x, w1q, s1, b1, h1, H);
  // fc2
  gemm_nt_kernel<1024, false, true,  true,  true ><<<B / 64, 256, 0, stream>>>(
      h1, w2q, s2, b2, h2, H);
  // heads: no quant scale, no relu, f32 out with col<1000 guard
  gemm_nt_kernel<1024, false, false, false, false><<<B / 64, 256, 0, stream>>>(
      h2, hwq, nullptr, hb, d_out, NC);
}
